// MultiHeadAttention_67791763800224
// MI455X (gfx1250) — compile-verified, matrix operand order checked by reading, not run on hardware
//
#include <hip/hip_runtime.h>

typedef __bf16 bf16_t;
typedef __attribute__((ext_vector_type(16))) __bf16 v16bf;
typedef __attribute__((ext_vector_type(8)))  __bf16 v8bf;
typedef __attribute__((ext_vector_type(8)))  float  v8f;

#define WMMA_BF16(A, B, C) \
  __builtin_amdgcn_wmma_f32_16x16x32_bf16(false, (A), false, (B), (short)0, (C), false, false)

// Async global -> LDS copy, 16 bytes per lane, tracked by ASYNCcnt.
// vdst = 32-bit LDS byte offset, vaddr = 64-bit global address.
__device__ __forceinline__ void async_copy_b128(unsigned lds_off, const void* gaddr) {
  asm volatile("global_load_async_to_lds_b128 %0, %1, off"
               :: "v"(lds_off), "v"(gaddr)
               : "memory");
}

// Load a 16x32 bf16 fragment (A-layout; identical pattern serves as B-frag when
// memory holds B^T row-major along K). ISA 7.12.2: lane = {r=lane&15, h=lane>>4},
// elements 0..7 <-> K = 8h+j, elements 8..15 <-> K = 16+8h+j.
__device__ __forceinline__ v16bf load_frag(const bf16_t* base, int stride, int lane) {
  const int r = lane & 15, hh = lane >> 4;
  const bf16_t* p = base + (size_t)r * stride + hh * 8;
  union { v16bf v; v8bf q[2]; } u;
  u.q[0] = *(const v8bf*)(p);
  u.q[1] = *(const v8bf*)(p + 16);
  return u.v;
}

// ---------------- elementwise conversion ----------------
__global__ void cvt_f32_bf16(const float* __restrict__ in, bf16_t* __restrict__ out, int n) {
  int i = blockIdx.x * blockDim.x + threadIdx.x;
  if (i < n) out[i] = (bf16_t)in[i];
}

// qp[b,s,1024] fp32 -> qh[b,h,s,64] bf16, scaled by 1/sqrt(64)
__global__ void pack_q(const float* __restrict__ qp, bf16_t* __restrict__ qh) {
  int i = blockIdx.x * blockDim.x + threadIdx.x;        // 0 .. 4M
  int dm = i & 1023; int bs = i >> 10;
  int s = bs & 2047; int b = bs >> 11;
  int head = dm >> 6, d = dm & 63;
  qh[(((size_t)(b * 16 + head) * 2048) + s) * 64 + d] = (bf16_t)(qp[i] * 0.125f);
}

// Interleaved RoPE over the full 1024-dim row (matches reference: d=1024, half=512).
// kp[b,s,1024] fp32 -> kh[b,h,s,64] bf16
__global__ void rope_pack_k(const float* __restrict__ kp, bf16_t* __restrict__ kh) {
  int i = blockIdx.x * blockDim.x + threadIdx.x;        // pair index, 0 .. 2M
  int pi = i & 511; int bs = i >> 9;
  int s = bs & 2047; int b = bs >> 11;
  float ang = (float)s * __expf(-(float)pi * (9.210340371976184f / 512.0f));
  float sn, cs; __sincosf(ang, &sn, &cs);
  const float* src = kp + ((size_t)(b * 2048 + s)) * 1024 + 2 * pi;
  float x1 = src[0], x2 = src[1];
  float r1 = x1 * cs - x2 * sn, r2 = x1 * sn + x2 * cs;
  int e = 2 * pi; int head = e >> 6; int d = e & 63;
  bf16_t* dst = kh + (((size_t)(b * 16 + head) * 2048) + s) * 64 + d;
  dst[0] = (bf16_t)r1; dst[1] = (bf16_t)r2;
}

// vp[b,s,1024] fp32 -> vT[b,h,64,s] bf16 (transposed so P@V is contiguous along K)
__global__ void rope_pack_vT(const float* __restrict__ vp, bf16_t* __restrict__ vT) {
  int i = blockIdx.x * blockDim.x + threadIdx.x;        // pair index, 0 .. 2M
  int pi = i & 511; int bs = i >> 9;
  int s = bs & 2047; int b = bs >> 11;
  float ang = (float)s * __expf(-(float)pi * (9.210340371976184f / 512.0f));
  float sn, cs; __sincosf(ang, &sn, &cs);
  const float* src = vp + ((size_t)(b * 2048 + s)) * 1024 + 2 * pi;
  float x1 = src[0], x2 = src[1];
  float r1 = x1 * cs - x2 * sn, r2 = x1 * sn + x2 * cs;
  int e = 2 * pi; int head = e >> 6; int d = e & 63;
  bf16_t* dst = vT + (((size_t)(b * 16 + head) * 64 + d) * 2048 + s);
  dst[0] = (bf16_t)r1; dst[2048] = (bf16_t)r2;
}

// ---------------- bf16 GEMM: C[m,n] = sum_k A[m,k]*B[n,k] (+bias), fp32 out ----
// Block = 256 threads (8 waves). Block tile 128(M) x 64(N); wave = one 16x64 strip.
// B tile (64x32 bf16) double-buffered in LDS via async global->LDS copies:
// fetch of tile i+1 overlaps WMMAs of tile i; s_wait_asynccnt 1 + barrier
// publishes tile i to all 8 waves.
__global__ void gemm_bf16_f32(const bf16_t* __restrict__ A, const bf16_t* __restrict__ B,
                              float* __restrict__ C, const float* __restrict__ bias,
                              int M, int N, int K) {
  __shared__ bf16_t Bt[2][64][32];
  const int row0 = blockIdx.x * 128;
  const int col0 = blockIdx.y * 64;
  const int lane = threadIdx.x & 31;
  const int wave = threadIdx.x >> 5;                    // 0..7
  const int wrow = row0 + wave * 16;

  // Per-thread staging slot: thread t copies B[col0 + t/4][ (t%4)*8 .. +8 ).
  const int t = threadIdx.x;
  const int r = t >> 2, seg = (t & 3) * 8;
  const unsigned lds0 = (unsigned)(size_t)&Bt[0][r][seg];
  const unsigned lds1 = (unsigned)(size_t)&Bt[1][r][seg];
  const bf16_t* gsrc = &B[(size_t)(col0 + r) * K + seg];

  v8f acc[4] = {};
  async_copy_b128(lds0, gsrc);                          // tile 0 in flight
  for (int k0 = 0; k0 < K; k0 += 32) {
    const int it = k0 >> 5;
    const unsigned nxt_lds = (it & 1) ? lds0 : lds1;
    const int k_next = (k0 + 32 < K) ? (k0 + 32) : k0;  // last iter: harmless dup
    async_copy_b128(nxt_lds, gsrc + k_next);            // prefetch tile i+1
    __builtin_prefetch(A + (size_t)wrow * K + k0 + 32, 0, 1);  // global_prefetch_b8
    asm volatile("s_wait_asynccnt 0x1" ::: "memory");   // tile i landed (this wave)
    __syncthreads();                                     // ... for all 8 waves
    v16bf a = load_frag(A + (size_t)wrow * K + k0, K, lane);
#pragma unroll
    for (int j = 0; j < 4; ++j) {
      v16bf bb = load_frag(&Bt[it & 1][j * 16][0], 32, lane);
      acc[j] = WMMA_BF16(a, bb, acc[j]);
    }
    __syncthreads();                                     // reads done before next overwrite
  }
  const int n = lane & 15, hh = lane >> 4;
#pragma unroll
  for (int j = 0; j < 4; ++j)
#pragma unroll
    for (int v = 0; v < 8; ++v) {
      int m = wrow + v + hh * 8;
      int c = col0 + j * 16 + n;
      float val = acc[j][v];
      if (bias) val += bias[c];
      C[(size_t)m * N + c] = val;
    }
}

// ---------------- attention: one block per (b*h, q-tile of 16) -----------------
// 128 threads = 4 waves. 16x2048 fp32 score panel in LDS (128 KB).
__global__ void attention_kernel(const bf16_t* __restrict__ qh, const bf16_t* __restrict__ kh,
                                 const bf16_t* __restrict__ vT, bf16_t* __restrict__ ctx) {
  __shared__ float S[16][2048];
  __shared__ float red[16][8];
  __shared__ float lrow[16];
  const int bh = blockIdx.x;                            // 0..31
  const int b = bh >> 4, head = bh & 15;
  const int q0 = blockIdx.y * 16;
  const int lane = threadIdx.x & 31;
  const int wave = threadIdx.x >> 5;                    // 0..3
  const bf16_t* qbase = qh + ((size_t)bh * 2048 + q0) * 64;
  const bf16_t* kbase = kh + (size_t)bh * 2048 * 64;
  const bf16_t* vbase = vT + (size_t)bh * 64 * 2048;

  // Q fragments for this 16-row tile (d = 0..31 and 32..63), reused for all k.
  v16bf a0 = load_frag(qbase, 64, lane);
  v16bf a1 = load_frag(qbase + 32, 64, lane);

  // Phase 1: scores. Wave w covers k-columns [w*512, w*512+512).
  for (int kt = 0; kt < 32; ++kt) {
    const int kk = wave * 512 + kt * 16;
    v16bf b0 = load_frag(kbase + (size_t)kk * 64, 64, lane);
    v16bf b1 = load_frag(kbase + (size_t)kk * 64 + 32, 64, lane);
    v8f acc = {};
    acc = WMMA_BF16(a0, b0, acc);
    acc = WMMA_BF16(a1, b1, acc);
    const int n = lane & 15, hh = lane >> 4;
#pragma unroll
    for (int v = 0; v < 8; ++v) S[v + hh * 8][kk + n] = acc[v];
  }
  __syncthreads();

  // Phase 2: softmax stats (8 threads per row, 256 columns each).
  {
    const int row = threadIdx.x >> 3, sub = threadIdx.x & 7;
    const int c0 = sub * 256;
    float m = -1e30f;
    for (int c = 0; c < 256; ++c) m = fmaxf(m, S[row][c0 + c]);
    red[row][sub] = m;
    __syncthreads();
    float rm = red[row][0];
#pragma unroll
    for (int j = 1; j < 8; ++j) rm = fmaxf(rm, red[row][j]);
    __syncthreads();
    float sum = 0.f;
    for (int c = 0; c < 256; ++c) {
      float e = __expf(S[row][c0 + c] - rm);
      S[row][c0 + c] = e;
      sum += e;
    }
    red[row][sub] = sum;
    __syncthreads();
    if (sub == 0) {
      float l = 0.f;
#pragma unroll
      for (int j = 0; j < 8; ++j) l += red[row][j];
      lrow[row] = l;
    }
  }
  __syncthreads();

  // Phase 3: O = P @ V. Wave w owns output d-tile w (16 cols). A-frags of P are
  // rebuilt from LDS fp32 -> bf16 (the D-layout -> A-layout transpose).
  {
    const int n = lane & 15, hh = lane >> 4;
    const int r = lane & 15;
    v8f acc = {};
    for (int kb = 0; kb < 2048; kb += 32) {
      const float* p = &S[r][kb + hh * 8];
      v16bf a;
#pragma unroll
      for (int j = 0; j < 8; ++j) a[j] = (bf16_t)p[j];
#pragma unroll
      for (int j = 0; j < 8; ++j) a[8 + j] = (bf16_t)p[16 + j];
      v16bf bb = load_frag(vbase + (size_t)(wave * 16) * 2048 + kb, 2048, lane);
      acc = WMMA_BF16(a, bb, acc);
    }
#pragma unroll
    for (int v = 0; v < 8; ++v) {
      int m = v + hh * 8;
      float o = acc[v] / lrow[m];
      ctx[((size_t)(b * 2048 + q0 + m)) * 1024 + head * 64 + wave * 16 + n] = (bf16_t)o;
    }
  }
}

// ---------------- host side ----------------
extern "C" void kernel_launch(void* const* d_in, const int* in_sizes, int n_in,
                              void* d_out, int out_size, void* d_ws, size_t ws_size,
                              hipStream_t stream) {
  (void)in_sizes; (void)n_in; (void)out_size; (void)ws_size;
  const float* q  = (const float*)d_in[0];
  const float* k  = (const float*)d_in[1];
  const float* v  = (const float*)d_in[2];
  const float* Wq = (const float*)d_in[3];
  const float* Wk = (const float*)d_in[4];
  const float* Wv = (const float*)d_in[5];
  const float* Wp = (const float*)d_in[6];
  const float* bp = (const float*)d_in[7];
  float* out = (float*)d_out;

  const int M = 2 * 2048;          // B*SEQ = 4096 rows
  const int D = 1024;
  const size_t MD = (size_t)M * D; // 4M elements
  const size_t WD = (size_t)D * D; // 1M elements

  char* w = (char*)d_ws;
  auto bump = [&](size_t bytes) { char* p = w; w += (bytes + 255) & ~(size_t)255; return p; };
  bf16_t* q16   = (bf16_t*)bump(MD * 2);
  bf16_t* k16   = (bf16_t*)bump(MD * 2);
  bf16_t* v16   = (bf16_t*)bump(MD * 2);
  bf16_t* Wq16  = (bf16_t*)bump(WD * 2);
  bf16_t* Wk16  = (bf16_t*)bump(WD * 2);
  bf16_t* Wv16  = (bf16_t*)bump(WD * 2);
  bf16_t* Wp16  = (bf16_t*)bump(WD * 2);
  float*  qp    = (float*)bump(MD * 4);
  float*  kp    = (float*)bump(MD * 4);
  float*  vp    = (float*)bump(MD * 4);
  bf16_t* qh    = (bf16_t*)bump(MD * 2);
  bf16_t* kh    = (bf16_t*)bump(MD * 2);
  bf16_t* vT    = (bf16_t*)bump(MD * 2);
  bf16_t* ctx16 = (bf16_t*)bump(MD * 2);

  const int TPB = 256;
  // 1) fp32 -> bf16
  cvt_f32_bf16<<<(int)(MD / TPB), TPB, 0, stream>>>(q,  q16,  (int)MD);
  cvt_f32_bf16<<<(int)(MD / TPB), TPB, 0, stream>>>(k,  k16,  (int)MD);
  cvt_f32_bf16<<<(int)(MD / TPB), TPB, 0, stream>>>(v,  v16,  (int)MD);
  cvt_f32_bf16<<<(int)(WD / TPB), TPB, 0, stream>>>(Wq, Wq16, (int)WD);
  cvt_f32_bf16<<<(int)(WD / TPB), TPB, 0, stream>>>(Wk, Wk16, (int)WD);
  cvt_f32_bf16<<<(int)(WD / TPB), TPB, 0, stream>>>(Wv, Wv16, (int)WD);
  cvt_f32_bf16<<<(int)(WD / TPB), TPB, 0, stream>>>(Wp, Wp16, (int)WD);

  // 2) projections: X @ W^T
  dim3 ggrid(M / 128, D / 64);
  gemm_bf16_f32<<<ggrid, 256, 0, stream>>>(q16, Wq16, qp, nullptr, M, D, D);
  gemm_bf16_f32<<<ggrid, 256, 0, stream>>>(k16, Wk16, kp, nullptr, M, D, D);
  gemm_bf16_f32<<<ggrid, 256, 0, stream>>>(v16, Wv16, vp, nullptr, M, D, D);

  // 3) rope (k, v) + head-major repack, q scale+repack
  pack_q      <<<(int)(MD / TPB),     TPB, 0, stream>>>(qp, qh);
  rope_pack_k <<<(int)(MD / 2 / TPB), TPB, 0, stream>>>(kp, kh);
  rope_pack_vT<<<(int)(MD / 2 / TPB), TPB, 0, stream>>>(vp, vT);

  // 4) attention
  attention_kernel<<<dim3(32, 128), 128, 0, stream>>>(qh, kh, vT, ctx16);

  // 5) output projection + bias
  gemm_bf16_f32<<<ggrid, 256, 0, stream>>>(ctx16, Wp16, out, bp, M, D, D);
}
